// MultiHeadAttention_83056077570808
// MI455X (gfx1250) — compile-verified, matrix operand order checked by reading, not run on hardware
//
#include <hip/hip_runtime.h>
#include <hip/hip_bf16.h>
#include <math.h>

// ---------------------------------------------------------------------------
// Types
// ---------------------------------------------------------------------------
typedef __attribute__((ext_vector_type(16))) __bf16 v16bf;
typedef __attribute__((ext_vector_type(8)))  __bf16 v8bf;
typedef __attribute__((ext_vector_type(8)))  float  v8f;

#define T_SEQ   2048
#define DMODEL  4096
#define NQH     32
#define NKVH    8
#define HD      128
#define ATTN_MULT 0.08838834764831845f
#define CAP       30.0f

// ---------------------------------------------------------------------------
// f32 -> bf16 conversion helpers (prefer HW packed cvt when declared)
// ---------------------------------------------------------------------------
__device__ __forceinline__ unsigned short bfbits(float x) {
    unsigned u = __builtin_bit_cast(unsigned, x);
    u += 0x7FFFu + ((u >> 16) & 1u);            // round-to-nearest-even
    return (unsigned short)(u >> 16);
}

#if __has_builtin(__builtin_amdgcn_cvt_pk_bf16_f32)
typedef __attribute__((ext_vector_type(2))) __bf16 v2bf;
__device__ __forceinline__ __bf16 f2bf(float x) {
    v2bf t = __builtin_amdgcn_cvt_pk_bf16_f32(x, x);
    return t[0];
}
__device__ __forceinline__ ushort4 f2bf4(float4 v) {
    v2bf lo = __builtin_amdgcn_cvt_pk_bf16_f32(v.x, v.y);
    v2bf hi = __builtin_amdgcn_cvt_pk_bf16_f32(v.z, v.w);
    uint2 u;
    u.x = __builtin_bit_cast(unsigned, lo);
    u.y = __builtin_bit_cast(unsigned, hi);
    return __builtin_bit_cast(ushort4, u);
}
#else
__device__ __forceinline__ __bf16 f2bf(float x) {
    unsigned short h = bfbits(x);
    return __builtin_bit_cast(__bf16, h);
}
__device__ __forceinline__ ushort4 f2bf4(float4 v) {
    ushort4 r; r.x = bfbits(v.x); r.y = bfbits(v.y); r.z = bfbits(v.z); r.w = bfbits(v.w);
    return r;
}
#endif

// Load a 16-bit A/B fragment half: elements {p[0..7], p[16..23]} (two b128 ops)
__device__ __forceinline__ v16bf load16(const __bf16* p) {
    v8bf lo = *(const v8bf*)p;
    v8bf hi = *(const v8bf*)(p + 16);
    v16bf r;
#pragma unroll
    for (int i = 0; i < 8; ++i) { r[i] = lo[i]; r[i + 8] = hi[i]; }
    return r;
}

__device__ __forceinline__ v8f wmma_bf16(v16bf a, v16bf b, v8f c) {
    return __builtin_amdgcn_wmma_f32_16x16x32_bf16(
        /*neg_a=*/false, a, /*neg_b=*/false, b,
        /*c_mod=*/(short)0, c, /*reuse_a=*/false, /*reuse_b=*/false);
}

// Async global -> LDS copies (CDNA5 GLOBAL_LOAD_ASYNC_TO_LDS_*, ASYNCcnt).
// LDS address = low 32 bits of generic pointer.
__device__ __forceinline__ void async_g2l_b128(const void* gptr, void* lptr) {
    unsigned loff = (unsigned)(uintptr_t)lptr;
    asm volatile("global_load_async_to_lds_b128 %0, %1, off"
                 :: "v"(loff), "v"(gptr) : "memory");
}
__device__ __forceinline__ void async_g2l_b64(const void* gptr, void* lptr) {
    unsigned loff = (unsigned)(uintptr_t)lptr;
    asm volatile("global_load_async_to_lds_b64 %0, %1, off"
                 :: "v"(loff), "v"(gptr) : "memory");
}
__device__ __forceinline__ void wait_async0() {
    asm volatile("s_wait_asynccnt 0x0" ::: "memory");
}

// Fast tanh for bounded inputs: tanh(x) = 1 - 2/(e^{2x}+1), single v_rcp_f32
__device__ __forceinline__ float fast_tanhf(float x) {
    float e = __expf(2.0f * x);
    return 1.0f - 2.0f * __builtin_amdgcn_rcpf(e + 1.0f);
}

// ---------------------------------------------------------------------------
// Generic WMMA GEMM: C[M,N] = A[M,K] * B[K,N], row-major. A is f32 or bf16,
// B is f32 (converted to bf16 on stage-in), C is f32.
// Block: 128 threads (4 waves). Tile: 128(M) x 64(N), k-step 32.
// Wave w computes rows [32w, 32w+32) of the tile (2 A-fragments, 8 WMMA/kstep).
// bf16 A tiles are staged with async global->LDS B64 copies.
// ---------------------------------------------------------------------------
template <typename TA>
__global__ __launch_bounds__(128)
void gemm_wmma(const TA* __restrict__ A, const float* __restrict__ B,
               float* __restrict__ C, int M, int N, int K) {
    __shared__ __bf16 As[128][48];  // row stride 96B (16B aligned)
    __shared__ __bf16 Bs[64][48];   // stored transposed: Bs[n][k]

    const int tid  = threadIdx.x;
    const int lane = tid & 31;
    const int wave = tid >> 5;
    const int l15  = lane & 15;
    const int hl   = lane >> 4;          // lane half
    const int m0   = blockIdx.x * 128;
    const int n0   = blockIdx.y * 64;

    v8f acc0[4] = {};
    v8f acc1[4] = {};

    for (int k0 = 0; k0 < K; k0 += 32) {
        // ---- stage A tile (128 x 32) ----
#pragma unroll
        for (int it = 0; it < 8; ++it) {
            int s  = tid + it * 128;       // 1024 quad-slots
            int r  = s >> 3;
            int c4 = (s & 7) * 4;
            const TA* ap = &A[(size_t)(m0 + r) * K + k0 + c4];
            if (k0 + 32 < K) __builtin_prefetch(ap + 32, 0, 0);
            if constexpr (sizeof(TA) == 4) {
                float4 v = *(const float4*)ap;
                *(ushort4*)&As[r][c4] = f2bf4(v);
            } else {
                async_g2l_b64(ap, &As[r][c4]);   // pure bf16 copy, ASYNCcnt
            }
        }
        // ---- stage B tile (32 x 64), transposed into Bs[n][k] ----
#pragma unroll
        for (int it = 0; it < 4; ++it) {
            int s   = tid + it * 128;
            int kr  = s >> 4;
            int nc4 = (s & 15) * 4;
            const float* bp = &B[(size_t)(k0 + kr) * N + n0 + nc4];
            if (k0 + 32 < K) __builtin_prefetch(bp + (size_t)32 * N, 0, 0);
            float4 v = *(const float4*)bp;
            Bs[nc4 + 0][kr] = f2bf(v.x);
            Bs[nc4 + 1][kr] = f2bf(v.y);
            Bs[nc4 + 2][kr] = f2bf(v.z);
            Bs[nc4 + 3][kr] = f2bf(v.w);
        }
        if constexpr (sizeof(TA) == 2) wait_async0();
        __syncthreads();

        // ---- fragments + WMMA: 2 A-frags x 4 B-frags = 8 WMMA ----
        v16bf a0 = load16(&As[wave * 32 + l15][hl * 8]);
        v16bf a1 = load16(&As[wave * 32 + 16 + l15][hl * 8]);
#pragma unroll
        for (int j = 0; j < 4; ++j) {
            v16bf b = load16(&Bs[j * 16 + l15][hl * 8]);
            acc0[j] = wmma_bf16(a0, b, acc0[j]);
            acc1[j] = wmma_bf16(a1, b, acc1[j]);
        }
        __syncthreads();
    }

    // ---- epilogue: C layout = VGPR r holds (M=r|r+8, N=lane%16) ----
#pragma unroll
    for (int j = 0; j < 4; ++j)
#pragma unroll
        for (int r = 0; r < 8; ++r) {
            int row = m0 + wave * 32 + r + hl * 8;
            int col = n0 + j * 16 + l15;
            C[(size_t)row * N + col]        = acc0[j][r];
            C[(size_t)(row + 16) * N + col] = acc1[j][r];
        }
}

// ---------------------------------------------------------------------------
// RoPE (half-rotation, phase tiled x2) + f32 -> bf16.
// X: [T, n_heads*128] f32, Y: [T, n_heads, 128] bf16. One thread per (t,h,i<64).
// ---------------------------------------------------------------------------
__global__ void rope_bf16_kernel(const float* __restrict__ X, __bf16* __restrict__ Y,
                                 int n_heads, int total) {
    int idx = blockIdx.x * blockDim.x + threadIdx.x;
    if (idx >= total) return;
    int i    = idx & 63;
    int th   = idx >> 6;           // t*n_heads + h
    int t    = th / n_heads;
    int base = th * 128;
    float x1 = X[base + i];
    float x2 = X[base + i + 64];
    // inv_freq = 10000^(-2i/128) = exp(-2i/128 * ln(10000))
    float freq = __expf(-(float)(2 * i) * (9.210340371976184f / 128.0f));
    float ph = (float)t * freq;
    float s, c;
    __sincosf(ph, &s, &c);
    Y[base + i]      = f2bf(x1 * c - x2 * s);
    Y[base + i + 64] = f2bf(x2 * c + x1 * s);
}

// f32 -> bf16 bulk convert (vectorized x4)
__global__ void cvt_bf16_kernel(const float* __restrict__ X, __bf16* __restrict__ Y, int n4) {
    int i = blockIdx.x * blockDim.x + threadIdx.x;
    if (i >= n4) return;
    float4 v = ((const float4*)X)[i];
    ((ushort4*)Y)[i] = f2bf4(v);
}

// ---------------------------------------------------------------------------
// Flash attention, causal, tanh softcap. GQA: q head h uses kv head h/4.
// Block = (64 q rows) x (1 q head): 128 threads = 4 waves, wave -> 16 q rows.
// K tile staged with GLOBAL_LOAD_ASYNC_TO_LDS_B128 (ASYNCcnt), V tile is
// transposed through VGPRs into LDS, P is relayouted C->A via per-wave LDS.
// ---------------------------------------------------------------------------
__global__ __launch_bounds__(128)
void attn_kernel(const __bf16* __restrict__ Q, const __bf16* __restrict__ K,
                 const __bf16* __restrict__ V, __bf16* __restrict__ O) {
    const int h   = blockIdx.y;
    const int kvh = h >> 2;
    const int t0  = blockIdx.x * 64;
    const int tid = threadIdx.x;
    const int lane = tid & 31, wave = tid >> 5;
    const int l15 = lane & 15, hl = lane >> 4;

    __shared__ __bf16 Ks[32][144];          // [key][hd], 288B stride
    __shared__ __bf16 Vt[128][48];          // [hd][key], 96B stride
    __shared__ __bf16 Ps[4][16][48];        // per-wave P tile (C->A relayout)

    // Q fragments for this wave's 16 rows (4 chunks of k=32)
    const int qrow = t0 + wave * 16 + l15;
    v16bf qf[4];
#pragma unroll
    for (int c = 0; c < 4; ++c)
        qf[c] = load16(&Q[((size_t)qrow * NQH + h) * HD + c * 32 + hl * 8]);

    float mrow[8], lrow[8];
    v8f o[8] = {};
#pragma unroll
    for (int r = 0; r < 8; ++r) { mrow[r] = -1e30f; lrow[r] = 0.f; }

    const int kend = t0 + 64;                     // causal upper bound (exclusive)
    for (int kb = 0; kb < kend; kb += 32) {
        // ---- stage K block (32 keys x 128 dims): async global->LDS b128 ----
#pragma unroll
        for (int it = 0; it < 4; ++it) {
            int s  = tid + it * 128;
            int kr = s >> 4;
            int c8 = (s & 15) * 8;
            async_g2l_b128(&K[((size_t)(kb + kr) * NKVH + kvh) * HD + c8],
                           &Ks[kr][c8]);
        }
        // ---- stage V block transposed: Vt[d][key] ----
#pragma unroll
        for (int it = 0; it < 4; ++it) {
            int s  = tid + it * 128;
            int kr = s >> 4;
            int c8 = (s & 15) * 8;
            const __bf16* vp = &V[((size_t)(kb + kr) * NKVH + kvh) * HD + c8];
            if (kb + 32 < kend) __builtin_prefetch(vp + 32 * NKVH * HD, 0, 0);
            v8bf v = *(const v8bf*)vp;
#pragma unroll
            for (int i = 0; i < 8; ++i) Vt[c8 + i][kr] = v[i];
        }
        wait_async0();                 // this wave's async K copies landed
        __syncthreads();               // everyone's K/V tiles visible

        // ---- S = Q . K^T : 16 x 32 in two 16x16 accumulators ----
        v8f sacc[2] = {};
#pragma unroll
        for (int c = 0; c < 4; ++c) {
            v16bf b0 = load16(&Ks[l15][c * 32 + hl * 8]);
            v16bf b1 = load16(&Ks[16 + l15][c * 32 + hl * 8]);
            sacc[0] = wmma_bf16(qf[c], b0, sacc[0]);
            sacc[1] = wmma_bf16(qf[c], b1, sacc[1]);
        }

        // ---- softcap + causal mask ----
        float pp[2][8];
#pragma unroll
        for (int j = 0; j < 2; ++j) {
            int key_abs = kb + j * 16 + l15;
#pragma unroll
            for (int r = 0; r < 8; ++r) {
                float s = CAP * fast_tanhf(sacc[j][r] * (ATTN_MULT / CAP));
                int row_abs = t0 + wave * 16 + r + hl * 8;
                if (key_abs > row_abs) s = -1e30f;
                pp[j][r] = s;
            }
        }

        // ---- online softmax: rows live in 16-lane halves ----
        float corr[8];
#pragma unroll
        for (int r = 0; r < 8; ++r) {
            float mx = fmaxf(pp[0][r], pp[1][r]);
#pragma unroll
            for (int off = 8; off >= 1; off >>= 1)
                mx = fmaxf(mx, __shfl_xor(mx, off, 32));
            float mnew = fmaxf(mrow[r], mx);
            float s0 = __expf(pp[0][r] - mnew);
            float s1 = __expf(pp[1][r] - mnew);
            pp[0][r] = s0; pp[1][r] = s1;
            float sm = s0 + s1;
#pragma unroll
            for (int off = 8; off >= 1; off >>= 1)
                sm += __shfl_xor(sm, off, 32);
            corr[r] = __expf(mrow[r] - mnew);
            lrow[r] = lrow[r] * corr[r] + sm;
            mrow[r] = mnew;
        }
#pragma unroll
        for (int j = 0; j < 8; ++j)
#pragma unroll
            for (int r = 0; r < 8; ++r) o[j][r] *= corr[r];

        // ---- relayout P (C layout) -> A layout via per-wave LDS ----
#pragma unroll
        for (int j = 0; j < 2; ++j)
#pragma unroll
            for (int r = 0; r < 8; ++r)
                Ps[wave][r + hl * 8][j * 16 + l15] = f2bf(pp[j][r]);
        v16bf pf = load16(&Ps[wave][l15][hl * 8]);

        // ---- O += P . V (8 d-subtiles, k=32 keys) ----
#pragma unroll
        for (int j = 0; j < 8; ++j) {
            v16bf vb = load16(&Vt[j * 16 + l15][hl * 8]);
            o[j] = wmma_bf16(pf, vb, o[j]);
        }
        __syncthreads();
    }

    // ---- epilogue: normalize and emit bf16 [t, h*128+d] ----
#pragma unroll
    for (int j = 0; j < 8; ++j)
#pragma unroll
        for (int r = 0; r < 8; ++r) {
            int row = t0 + wave * 16 + r + hl * 8;
            int col = h * HD + j * 16 + l15;
            O[(size_t)row * DMODEL + col] = f2bf(o[j][r] / lrow[r]);
        }
}

// ---------------------------------------------------------------------------
// Host launcher
// ---------------------------------------------------------------------------
extern "C" void kernel_launch(void* const* d_in, const int* in_sizes, int n_in,
                              void* d_out, int out_size, void* d_ws, size_t ws_size,
                              hipStream_t stream) {
    (void)in_sizes; (void)n_in; (void)out_size; (void)ws_size;
    const float* Xq = (const float*)d_in[0];
    const float* Xk = (const float*)d_in[1];
    const float* Xv = (const float*)d_in[2];
    // d_in[3] = mask (causal, known analytically — unused)
    const float* Wq = (const float*)d_in[4];
    const float* Wk = (const float*)d_in[5];
    const float* Wv = (const float*)d_in[6];
    const float* Wo = (const float*)d_in[7];
    float* out = (float*)d_out;

    char* ws = (char*)d_ws;
    float*  Qf = (float*)ws;  ws += (size_t)T_SEQ * DMODEL * 4;         // 32 MB
    float*  Kf = (float*)ws;  ws += (size_t)T_SEQ * NKVH * HD * 4;      //  8 MB
    float*  Vf = (float*)ws;  ws += (size_t)T_SEQ * NKVH * HD * 4;      //  8 MB
    __bf16* Qb = (__bf16*)ws; ws += (size_t)T_SEQ * DMODEL * 2;         // 16 MB
    __bf16* Kb = (__bf16*)ws; ws += (size_t)T_SEQ * NKVH * HD * 2;      //  4 MB
    __bf16* Vb = (__bf16*)ws; ws += (size_t)T_SEQ * NKVH * HD * 2;      //  4 MB
    __bf16* Ab = (__bf16*)ws;                                           // 16 MB

    dim3 blk(128);
    // QKV projections (f32 inputs, bf16 WMMA, f32 outputs)
    gemm_wmma<float><<<dim3(T_SEQ / 128, DMODEL / 64), blk, 0, stream>>>(Xq, Wq, Qf, T_SEQ, DMODEL, DMODEL);
    gemm_wmma<float><<<dim3(T_SEQ / 128, (NKVH * HD) / 64), blk, 0, stream>>>(Xk, Wk, Kf, T_SEQ, NKVH * HD, DMODEL);
    gemm_wmma<float><<<dim3(T_SEQ / 128, (NKVH * HD) / 64), blk, 0, stream>>>(Xv, Wv, Vf, T_SEQ, NKVH * HD, DMODEL);

    // RoPE -> bf16 for Q and K; plain convert for V
    int totq = T_SEQ * NQH * 64;
    rope_bf16_kernel<<<(totq + 255) / 256, 256, 0, stream>>>(Qf, Qb, NQH, totq);
    int totk = T_SEQ * NKVH * 64;
    rope_bf16_kernel<<<(totk + 255) / 256, 256, 0, stream>>>(Kf, Kb, NKVH, totk);
    int nv4 = (T_SEQ * NKVH * HD) / 4;
    cvt_bf16_kernel<<<(nv4 + 255) / 256, 256, 0, stream>>>(Vf, Vb, nv4);

    // Flash attention (causal + softcap), bf16 in/out
    attn_kernel<<<dim3(T_SEQ / 64, NQH), blk, 0, stream>>>(Qb, Kb, Vb, Ab);

    // Output projection: attn(bf16) @ Wo(f32) -> f32
    gemm_wmma<__bf16><<<dim3(T_SEQ / 128, DMODEL / 64), blk, 0, stream>>>(Ab, Wo, out, T_SEQ, DMODEL, DMODEL);
}
